// DifferentialAttention_89283780149865
// MI455X (gfx1250) — matrix operand order repaired and checked by reading, not compiled, not run on hardware
//
#include <hip/hip_runtime.h>

typedef _Float16 f16;
typedef __attribute__((ext_vector_type(16))) _Float16 v16h;
typedef __attribute__((ext_vector_type(8)))  _Float16 v8h;
typedef __attribute__((ext_vector_type(8)))  float    v8f;

#define WMMA_F16(a, b, c) \
    __builtin_amdgcn_wmma_f32_16x16x32_f16(false, (a), false, (b), (short)0, (c), false, false)

constexpr int kBATCH = 4;
constexpr int kS     = 4096;
constexpr int kD     = 1024;   // dim_model
constexpr int kE     = 128;    // 2*head_dim
constexpr float kScale = 0.125f; // 64^-0.5

// ---- workspace layout (bytes) ----
constexpr size_t WS_LAM = 0;                                    // 1 float
constexpr size_t WS_W   = 256;                                  // 3 * 128*1024 f16 = 768KB
constexpr size_t WS_Q   = WS_W + 3ull * kE * kD * sizeof(f16);  // 16384*128 f16 (pre-scaled)
constexpr size_t WS_K   = WS_Q + (size_t)kBATCH * kS * kE * sizeof(f16);
constexpr size_t WS_VT  = WS_K + (size_t)kBATCH * kS * kE * sizeof(f16); // V transposed [B][128][S]

// ---------------- lambda scalar ----------------
__global__ void dal_lambda_kernel(const float* lq1, const float* lq2,
                                  const float* lk1, const float* lk2, float* lam) {
    __shared__ float red[4];
    int t = threadIdx.x;              // 64 threads
    float p1 = lq1[t] * lk1[t];
    float p2 = lq2[t] * lk2[t];
    #pragma unroll
    for (int off = 16; off > 0; off >>= 1) {
        p1 += __shfl_down(p1, off, 32);
        p2 += __shfl_down(p2, off, 32);
    }
    if ((t & 31) == 0) { red[(t >> 5) * 2] = p1; red[(t >> 5) * 2 + 1] = p2; }
    __syncthreads();
    if (t == 0) {
        float s1 = red[0] + red[2];
        float s2 = red[1] + red[3];
        float lambda_init = 0.8f - 0.6f * __expf(-0.3f * 12.0f);
        *lam = __expf(s1) - __expf(s2) + lambda_init;
    }
}

// ---------------- weight f32 -> f16 ----------------
__global__ void dal_convw_kernel(const float* Wq, const float* Wk, const float* Wv, f16* Wh) {
    int i = blockIdx.x * 256 + threadIdx.x;  // grid covers 3*131072
    const int n = kE * kD;
    if (i < n)            Wh[i] = (f16)Wq[i];
    else if (i < 2 * n)   Wh[i] = (f16)Wk[i - n];
    else if (i < 3 * n)   Wh[i] = (f16)Wv[i - 2 * n];
}

// ---------------- projections: Qh (scaled), Kh, Vt ----------------
__global__ __launch_bounds__(256) void dal_proj_kernel(const float* __restrict__ x,
                                                       const f16* __restrict__ Wh,
                                                       f16* __restrict__ Qh,
                                                       f16* __restrict__ Kh,
                                                       f16* __restrict__ Vt) {
    __shared__ f16 xs[16 * kD];  // 32 KB
    const int t = threadIdx.x;
    const int mbase = blockIdx.x * 16;

    // load 16x1024 f32 tile of x, convert to f16 in LDS
    const float4* xg = (const float4*)(x + (size_t)mbase * kD);
    #pragma unroll
    for (int i = 0; i < 16; ++i) {
        int li = t + 256 * i;     // 0..4095 float4s
        float4 v = xg[li];
        int o = li * 4;
        xs[o + 0] = (f16)v.x; xs[o + 1] = (f16)v.y;
        xs[o + 2] = (f16)v.z; xs[o + 3] = (f16)v.w;
    }
    __syncthreads();

    const int wave = t >> 5;
    const int lane = t & 31;
    const int nb   = wave * 16;      // output-column tile base (8 waves x 16 = 128)
    const int r    = lane & 15;
    const int hg   = lane >> 4;      // lane half-group

    for (int p = 0; p < 3; ++p) {
        const f16* W = Wh + (size_t)p * kE * kD;
        v8f acc = {0.f, 0.f, 0.f, 0.f, 0.f, 0.f, 0.f, 0.f};
        for (int kt = 0; kt < kD; kt += 32) {
            // A fragment (16x32 f16): two contiguous 8-f16 chunks per lane
            const f16* ap = xs + r * kD + kt + (hg << 4);
            v8h lo = *(const v8h*)ap;
            v8h hi = *(const v8h*)(ap + 8);
            v16h a = __builtin_shufflevector(lo, hi, 0,1,2,3,4,5,6,7,8,9,10,11,12,13,14,15);
            // B fragment (32x16): 16 contiguous f16 of out-feature row nb+r
            const f16* bp = W + (size_t)(nb + r) * kD + kt + (hg << 4);
            v16h b = *(const v16h*)bp;
            acc = WMMA_F16(a, b, acc);
        }
        #pragma unroll
        for (int v = 0; v < 8; ++v) {
            int grow = mbase + v + (hg << 3);   // D layout: row = v + 8*hg
            int col  = nb + r;                  //           col = lane&15
            float val = acc[v];
            if (p == 0) {
                Qh[(size_t)grow * kE + col] = (f16)(val * kScale);
            } else if (p == 1) {
                Kh[(size_t)grow * kE + col] = (f16)val;
            } else {
                int bb = grow >> 12, ss = grow & (kS - 1);
                Vt[((size_t)bb * kE + col) * kS + ss] = (f16)val;
            }
        }
    }
}

// ---------------- fused differential flash attention ----------------
__global__ __launch_bounds__(128) void dal_attn_kernel(const f16* __restrict__ Qh,
                                                       const f16* __restrict__ Kh,
                                                       const f16* __restrict__ Vt,
                                                       const float* __restrict__ lamp,
                                                       float* __restrict__ out,
                                                       float* __restrict__ comb) {
    __shared__ f16 ktile[4][2][16 * kE];  // per-wave double-buffered K tile, 32 KB
    __shared__ f16 pbuf[4][16 * 32];      // per-wave P staging (D -> A relayout), 4 KB
    const int t    = threadIdx.x;
    const int wave = t >> 5;
    const int lane = t & 31;
    const int r    = lane & 15;
    const int hg   = lane >> 4;
    const int qt   = blockIdx.x * 4 + wave;   // 16-row q tile
    const int qrow = qt * 16;
    const int b    = qrow >> 12;
    const int sq   = qrow & (kS - 1);
    const float lam = lamp[0];

    // Q fragments: Q1 at cols 0..63, Q2 at cols 64..127 (scale pre-folded)
    const f16* qbase = Qh + (size_t)qrow * kE;
    auto ldA = [&](int kstep) -> v16h {
        const f16* p = qbase + r * kE + kstep + (hg << 4);
        v8h lo = *(const v8h*)p;
        v8h hi = *(const v8h*)(p + 8);
        return __builtin_shufflevector(lo, hi, 0,1,2,3,4,5,6,7,8,9,10,11,12,13,14,15);
    };
    const v16h q1a = ldA(0),  q1b = ldA(32);
    const v16h q2a = ldA(64), q2b = ldA(96);

    const f16* kmat = Kh + (size_t)b * kS * kE;
    const unsigned klds0 = (unsigned)(size_t)&ktile[wave][0][0];

    // async-copy one 4KB K tile (16 keys x 128 f16, contiguous) into LDS buffer `buf`
    auto stage_k = [&](int kt, int buf) {
        const char* src = (const char*)(kmat + (size_t)kt * 16 * kE);
        unsigned dst = klds0 + (unsigned)buf * (16 * kE * 2);
        #pragma unroll
        for (int j = 0; j < 8; ++j) {
            unsigned off = (unsigned)(lane + j * 32) * 16;   // 256 x 16B chunks
            unsigned l = dst + off;
            const char* g = src + off;
            asm volatile("global_load_async_to_lds_b128 %0, %1, off"
                         :: "v"(l), "v"(g) : "memory");
        }
    };

    auto scores = [&](int buf, v8f& s1, v8f& s2) {
        const f16* kc = &ktile[wave][buf][r * kE + (hg << 4)];
        v16h k1a = *(const v16h*)(kc);
        v16h k1b = *(const v16h*)(kc + 32);
        v16h k2a = *(const v16h*)(kc + 64);
        v16h k2b = *(const v16h*)(kc + 96);
        v8f z = {0.f, 0.f, 0.f, 0.f, 0.f, 0.f, 0.f, 0.f};
        s1 = WMMA_F16(q1b, k1b, WMMA_F16(q1a, k1a, z));
        s2 = WMMA_F16(q2b, k2b, WMMA_F16(q2a, k2a, z));
    };

    // ---- sweep 1: online max/sum for both score matrices ----
    float m1[8], l1[8], m2[8], l2[8];
    #pragma unroll
    for (int v = 0; v < 8; ++v) { m1[v] = m2[v] = -1e30f; l1[v] = l2[v] = 0.f; }

    stage_k(0, 0);
    for (int kt = 0; kt < kS / 16; ++kt) {
        if (kt + 1 < kS / 16) {
            stage_k(kt + 1, (kt + 1) & 1);
            asm volatile("s_wait_asynccnt 0x8" ::: "memory");   // tile kt resident
        } else {
            asm volatile("s_wait_asynccnt 0x0" ::: "memory");
        }
        v8f s1, s2;
        scores(kt & 1, s1, s2);
        #pragma unroll
        for (int v = 0; v < 8; ++v) {
            float a1 = s1[v], a2 = s2[v];
            float mx1 = a1, mx2 = a2;
            #pragma unroll
            for (int off = 1; off < 16; off <<= 1) {
                mx1 = fmaxf(mx1, __shfl_xor(mx1, off, 32));
                mx2 = fmaxf(mx2, __shfl_xor(mx2, off, 32));
            }
            float nm1 = fmaxf(m1[v], mx1);
            float nm2 = fmaxf(m2[v], mx2);
            float e1 = __expf(a1 - nm1);
            float e2 = __expf(a2 - nm2);
            #pragma unroll
            for (int off = 1; off < 16; off <<= 1) {
                e1 += __shfl_xor(e1, off, 32);
                e2 += __shfl_xor(e2, off, 32);
            }
            l1[v] = l1[v] * __expf(m1[v] - nm1) + e1;  m1[v] = nm1;
            l2[v] = l2[v] * __expf(m2[v] - nm2) + e2;  m2[v] = nm2;
        }
    }
    float il1[8], il2[8];
    #pragma unroll
    for (int v = 0; v < 8; ++v) { il1[v] = 1.f / l1[v]; il2[v] = 1.f / l2[v]; }

    // ---- sweep 2: normalized combined attn -> d_out, PV accumulate ----
    v8f o[8];
    #pragma unroll
    for (int c = 0; c < 8; ++c) o[c] = (v8f){0.f, 0.f, 0.f, 0.f, 0.f, 0.f, 0.f, 0.f};

    f16* P = &pbuf[wave][0];
    float* combB = comb + ((size_t)(b * kS + sq)) * kS;
    const f16* vmat = Vt + (size_t)b * kE * kS;

    stage_k(0, 0);
    for (int kp = 0; kp < kS / 32; ++kp) {
        #pragma unroll
        for (int half = 0; half < 2; ++half) {
            int kt = kp * 2 + half;
            if (kt + 1 < kS / 16) {
                stage_k(kt + 1, (kt + 1) & 1);
                asm volatile("s_wait_asynccnt 0x8" ::: "memory");
            } else {
                asm volatile("s_wait_asynccnt 0x0" ::: "memory");
            }
            v8f s1, s2;
            scores(kt & 1, s1, s2);
            #pragma unroll
            for (int v = 0; v < 8; ++v) {
                float p1 = __expf(s1[v] - m1[v]) * il1[v];
                float p2 = __expf(s2[v] - m2[v]) * il2[v];
                float cb = p1 - lam * p2;
                int row = v + (hg << 3);
                combB[(size_t)row * kS + kt * 16 + r] = cb;        // combined_attn out
                P[row * 32 + half * 16 + r] = (f16)cb;             // stage for PV
            }
        }
        __builtin_amdgcn_wave_barrier();   // keep P loads below P stores
        // A fragment of P (16 rows x 32 keys) from LDS
        const f16* pp = P + r * 32 + (hg << 4);
        v8h plo = *(const v8h*)pp;
        v8h phi = *(const v8h*)(pp + 8);
        v16h pa = __builtin_shufflevector(plo, phi, 0,1,2,3,4,5,6,7,8,9,10,11,12,13,14,15);
        #pragma unroll
        for (int c = 0; c < 8; ++c) {
            const f16* vp = vmat + (size_t)(c * 16 + r) * kS + kp * 32 + (hg << 4);
            v16h bfrag = *(const v16h*)vp;     // 32x16 slice of V (contiguous: V stored transposed)
            o[c] = WMMA_F16(pa, bfrag, o[c]);
        }
    }

    // epilogue: O (16 x 128)
    float* ob = out + ((size_t)(b * kS + sq)) * kE;
    #pragma unroll
    for (int c = 0; c < 8; ++c)
        #pragma unroll
        for (int v = 0; v < 8; ++v)
            ob[(size_t)(v + (hg << 3)) * kE + c * 16 + r] = o[c][v];
}

extern "C" void kernel_launch(void* const* d_in, const int* in_sizes, int n_in,
                              void* d_out, int out_size, void* d_ws, size_t ws_size,
                              hipStream_t stream) {
    const float* x   = (const float*)d_in[0];
    const float* Wq  = (const float*)d_in[1];
    const float* Wk  = (const float*)d_in[2];
    const float* Wv  = (const float*)d_in[3];
    const float* lq1 = (const float*)d_in[4];
    const float* lq2 = (const float*)d_in[5];
    const float* lk1 = (const float*)d_in[6];
    const float* lk2 = (const float*)d_in[7];

    char* ws = (char*)d_ws;
    float* lam = (float*)(ws + WS_LAM);
    f16*   Wh  = (f16*)(ws + WS_W);
    f16*   Qh  = (f16*)(ws + WS_Q);
    f16*   Kh  = (f16*)(ws + WS_K);
    f16*   Vt  = (f16*)(ws + WS_VT);

    float* out  = (float*)d_out;                        // [4,4096,128]
    float* comb = out + (size_t)kBATCH * kS * kE;       // [4,4096,4096]

    hipLaunchKernelGGL(dal_lambda_kernel, dim3(1), dim3(64), 0, stream, lq1, lq2, lk1, lk2, lam);
    hipLaunchKernelGGL(dal_convw_kernel, dim3((3 * kE * kD + 255) / 256), dim3(256), 0, stream,
                       Wq, Wk, Wv, Wh);
    hipLaunchKernelGGL(dal_proj_kernel, dim3(kBATCH * kS / 16), dim3(256), 0, stream,
                       x, Wh, Qh, Kh, Vt);
    hipLaunchKernelGGL(dal_attn_kernel, dim3(kBATCH * kS / 64), dim3(128), 0, stream,
                       Qh, Kh, Vt, lam, out, comb);
}
